// WanTransformerBlock_39874476376182
// MI455X (gfx1250) — compile-verified, matrix operand order checked by reading, not run on hardware
//
#include <hip/hip_runtime.h>
#include <hip/hip_bf16.h>
#include <math.h>

// ---------------------------------------------------------------------------
// CDNA5 (gfx1250) WanTransformerBlock: all GEMMs via v_wmma_f32_16x16x32_bf16
// Register-level software pipeline (global prefetch -> LDS double buffer),
// templated transB, 128x256 block tile, 64x64 wave tile (16 WMMA / K-step).
// ---------------------------------------------------------------------------

typedef __bf16 bf16;
typedef __attribute__((ext_vector_type(16))) __bf16 v16bf;
typedef __attribute__((ext_vector_type(8)))  __bf16 v8bf;
typedef __attribute__((ext_vector_type(8)))  float  v8f;

#define LDS_STRIDE 48   // bf16 elements per LDS row (96B, 16B aligned, padded)

// ---------------------------------------------------------------------------
// Tiled WMMA GEMM: C[M,N] = A[M,K](bf16) * B(bf16) (+bias) (epilogue)
//   A row-major (M x K). B: if TRANSB, row-major (N x K); else row-major (K x N).
//   mode 0: outF = v + bias        (fp32)
//   mode 1: outB = bf16(v + bias)
//   mode 2: outB = bf16(gelu(v + bias))   (exact GELU)
//   mode 3: outF += v + bias       (fp32 accumulate, residual)
// Block tile 128(M) x 256(N), 256 threads = 8 waves (2 along M x 4 along N),
// each wave computes 64x64 via 4x4 grid of 16x16x32 WMMA tiles.
// Pipeline per K-step: regs->LDS store, barrier, issue next global loads,
// then 8 fragment ds_loads + 16 WMMAs (loadcnt wait deferred a full tile).
// ---------------------------------------------------------------------------
template <int TRANSB>
__global__ __launch_bounds__(256)
void k_gemm_bf16_wmma(const bf16* __restrict__ A, int lda, long long strideA,
                      const bf16* __restrict__ Bm, int ldb, long long strideB,
                      const float* __restrict__ bias,
                      float* __restrict__ outF, bf16* __restrict__ outB,
                      int ldc, long long strideC,
                      int Ktot, int mode)
{
    __shared__ __align__(16) bf16 As[2][128 * LDS_STRIDE];
    __shared__ __align__(16) bf16 Bs[2][256 * LDS_STRIDE];

    const int bz = blockIdx.z;
    A  += (long long)bz * strideA;
    Bm += (long long)bz * strideB;
    const long long cOff = (long long)bz * strideC;

    const int tid   = threadIdx.x;
    const int lane  = tid & 31;
    const int wave  = tid >> 5;
    const int waveM = wave >> 2;   // 0..1
    const int waveN = wave & 3;    // 0..3

    const int m0 = blockIdx.y * 128;
    const int n0 = blockIdx.x * 256;

    // ---- per-thread staging addresses (hoisted; advanced by constants) ----
    // A tile 128x32: 512 chunks of 8 bf16; thread owns chunks t, t+256.
    const int am   = tid >> 2;           // row 0..63 (second chunk: +64)
    const int akc  = (tid & 3) << 3;     // k offset 0,8,16,24
    const bf16* aSrc = A + (long long)(m0 + am) * lda + akc;
    const long long aRow64 = (long long)64 * lda;
    const int aDst0 = am * LDS_STRIDE + akc;
    const int aDst1 = aDst0 + 64 * LDS_STRIDE;

    // B tile: TRANSB -> 256 rows x 32 (4 chunks/thread, rows +64 apart)
    //         else   -> 32 rows x 256 (4 chunks/thread, k rows +8 apart)
    const bf16* bSrc;
    long long bChunk, bStep;
    int bDst0 = 0, bkr = 0, bnc = 0;
    if (TRANSB) {
        bSrc   = Bm + (long long)(n0 + am) * ldb + akc;
        bChunk = (long long)64 * ldb;          // chunk-to-chunk stride
        bStep  = 32;                           // per K-step advance
        bDst0  = am * LDS_STRIDE + akc;
    } else {
        bkr = tid >> 5;                        // k row 0..7 (chunks: +8)
        bnc = (tid & 31) << 3;                 // n offset 0..248
        bSrc   = Bm + (long long)bkr * ldb + n0 + bnc;
        bChunk = (long long)8 * ldb;
        bStep  = (long long)32 * ldb;
    }

    // ---- register prefetch buffers ----
    v8bf ra0, ra1, rb[4];
    auto gload = [&]() {
        ra0 = *(const v8bf*)aSrc;
        ra1 = *(const v8bf*)(aSrc + aRow64);
#pragma unroll
        for (int q = 0; q < 4; ++q)
            rb[q] = *(const v8bf*)(bSrc + (long long)q * bChunk);
        aSrc += 32;
        bSrc += bStep;
    };
    auto lstore = [&](int buf) {
        *(v8bf*)&As[buf][aDst0] = ra0;
        *(v8bf*)&As[buf][aDst1] = ra1;
        if (TRANSB) {
#pragma unroll
            for (int q = 0; q < 4; ++q)
                *(v8bf*)&Bs[buf][bDst0 + q * 64 * LDS_STRIDE] = rb[q];
        } else {
#pragma unroll
            for (int q = 0; q < 4; ++q)
#pragma unroll
                for (int e = 0; e < 8; ++e)
                    Bs[buf][(bnc + e) * LDS_STRIDE + bkr + 8 * q] = rb[q][e];
        }
    };

    v8f acc[4][4];
#pragma unroll
    for (int i = 0; i < 4; ++i)
#pragma unroll
        for (int j = 0; j < 4; ++j)
#pragma unroll
            for (int e = 0; e < 8; ++e)
                acc[i][j][e] = 0.0f;

    // fragment base offsets (ISA 7.12.2 16-bit layout)
    const int hi = lane >> 4;     // half-wave
    const int r  = lane & 15;     // fragment row (A: m, B: n)
    const int aBase = (waveM * 64 + r) * LDS_STRIDE + hi * 8;
    const int bBase = (waveN * 64 + r) * LDS_STRIDE + hi * 8;

    // ---- prologue: first tile into registers ----
    gload();

    int p = 0;
    for (int k0 = 0; k0 < Ktot; k0 += 32) {
        // drain prefetch regs into LDS buffer p
        lstore(p);
        __syncthreads();
        // issue next tile's global loads (wait lands at next lstore)
        if (k0 + 32 < Ktot) gload();

        v16bf afrag[4];
#pragma unroll
        for (int i = 0; i < 4; ++i) {
            const bf16* q = &As[p][aBase + i * 16 * LDS_STRIDE];
            union { v16bf v; v8bf h[2]; } u;
            u.h[0] = *(const v8bf*)q;          // K 0..7   (+8 if hi)
            u.h[1] = *(const v8bf*)(q + 16);   // K 16..23 (+8 if hi)
            afrag[i] = u.v;
        }
#pragma unroll
        for (int j = 0; j < 4; ++j) {
            const bf16* q = &Bs[p][bBase + j * 16 * LDS_STRIDE];
            union { v16bf v; v8bf h[2]; } u;
            u.h[0] = *(const v8bf*)q;
            u.h[1] = *(const v8bf*)(q + 16);
            const v16bf bfrag = u.v;
#pragma unroll
            for (int i = 0; i < 4; ++i)
                acc[i][j] = __builtin_amdgcn_wmma_f32_16x16x32_bf16(
                    false, afrag[i], false, bfrag,
                    (short)0, acc[i][j], false, false);
        }
        p ^= 1;
    }
    __syncthreads();

    // ---- epilogue: D layout lane=(N%16), VGPR rr -> M = rr + 8*(lane>=16)
#pragma unroll
    for (int i = 0; i < 4; ++i) {
#pragma unroll
        for (int j = 0; j < 4; ++j) {
            const int col = n0 + waveN * 64 + j * 16 + r;
            const float bv = bias ? bias[col] : 0.0f;
#pragma unroll
            for (int rr = 0; rr < 8; ++rr) {
                const int row = m0 + waveM * 64 + i * 16 + hi * 8 + rr;
                float v = acc[i][j][rr] + bv;
                const long long idx = cOff + (long long)row * ldc + col;
                if (mode == 0) {
                    outF[idx] = v;
                } else if (mode == 1) {
                    outB[idx] = (bf16)v;
                } else if (mode == 2) {
                    outB[idx] = (bf16)(0.5f * v * (1.0f + erff(v * 0.70710678118654752f)));
                } else {
                    outF[idx] += v;
                }
            }
        }
    }
}

// ---------------------------------------------------------------------------
// Row kernels (one 256-thread block per row)
// ---------------------------------------------------------------------------
__global__ __launch_bounds__(256)
void k_rmsnorm_bf16(const float* __restrict__ x, const float* __restrict__ w,
                    bf16* __restrict__ out, int D)
{
    __shared__ float red[256];
    const float* row  = x   + (long long)blockIdx.x * D;
    bf16*        orow = out + (long long)blockIdx.x * D;
    const int tid = threadIdx.x;
    float s = 0.0f;
    for (int i = tid; i < D; i += 256) { float v = row[i]; s += v * v; }
    red[tid] = s; __syncthreads();
    for (int off = 128; off > 0; off >>= 1) {
        if (tid < off) red[tid] += red[tid + off];
        __syncthreads();
    }
    const float inv = rsqrtf(red[0] / (float)D + 1e-7f);
    for (int i = tid; i < D; i += 256) orow[i] = (bf16)(row[i] * inv * w[i]);
}

__global__ __launch_bounds__(256)
void k_layernorm_bf16(const float* __restrict__ x, const float* __restrict__ w,
                      const float* __restrict__ b, bf16* __restrict__ out, int D)
{
    __shared__ float r1[256];
    __shared__ float r2[256];
    const float* row  = x   + (long long)blockIdx.x * D;
    bf16*        orow = out + (long long)blockIdx.x * D;
    const int tid = threadIdx.x;
    float s = 0.0f, s2 = 0.0f;
    for (int i = tid; i < D; i += 256) { float v = row[i]; s += v; s2 += v * v; }
    r1[tid] = s; r2[tid] = s2; __syncthreads();
    for (int off = 128; off > 0; off >>= 1) {
        if (tid < off) { r1[tid] += r1[tid + off]; r2[tid] += r2[tid + off]; }
        __syncthreads();
    }
    const float mu  = r1[0] / (float)D;
    const float var = r2[0] / (float)D - mu * mu;
    const float inv = rsqrtf(var + 1e-5f);
    for (int i = tid; i < D; i += 256)
        orow[i] = (bf16)((row[i] - mu) * inv * w[i] + b[i]);
}

__global__ __launch_bounds__(256)
void k_softmax_bf16(const float* __restrict__ S, bf16* __restrict__ P,
                    int L, float scale)
{
    __shared__ float red[256];
    const long long rowIdx = (long long)blockIdx.y * gridDim.x + blockIdx.x;
    const float* row  = S + rowIdx * L;
    bf16*        orow = P + rowIdx * L;
    const int tid = threadIdx.x;

    float m = -3.0e38f;
    for (int i = tid; i < L; i += 256) m = fmaxf(m, row[i] * scale);
    red[tid] = m; __syncthreads();
    for (int off = 128; off > 0; off >>= 1) {
        if (tid < off) red[tid] = fmaxf(red[tid], red[tid + off]);
        __syncthreads();
    }
    m = red[0]; __syncthreads();

    float s = 0.0f;
    for (int i = tid; i < L; i += 256) s += __expf(row[i] * scale - m);
    red[tid] = s; __syncthreads();
    for (int off = 128; off > 0; off >>= 1) {
        if (tid < off) red[tid] += red[tid + off];
        __syncthreads();
    }
    const float inv = 1.0f / red[0];
    for (int i = tid; i < L; i += 256)
        orow[i] = (bf16)(__expf(row[i] * scale - m) * inv);
}

// ---------------------------------------------------------------------------
// Elementwise helpers
// ---------------------------------------------------------------------------
__global__ void k_cast_bf16(const float* __restrict__ in, bf16* __restrict__ out,
                            long long n)
{
    long long i = (long long)blockIdx.x * blockDim.x + threadIdx.x;
    const long long stride = (long long)gridDim.x * blockDim.x;
    for (; i < n; i += stride) out[i] = (bf16)in[i];
}

// in: K x N (fp32 row-major) -> out: N x K (bf16 row-major)
__global__ void k_cast_transpose_bf16(const float* __restrict__ in,
                                      bf16* __restrict__ out, int K, int N)
{
    long long idx = (long long)blockIdx.x * blockDim.x + threadIdx.x;
    const long long total  = (long long)K * N;
    const long long stride = (long long)gridDim.x * blockDim.x;
    for (; idx < total; idx += stride) {
        const int k = (int)(idx / N);
        const int n = (int)(idx - (long long)k * N);
        out[(long long)n * K + k] = (bf16)in[idx];
    }
}

__global__ void k_copy_f32(const float4* __restrict__ in, float4* __restrict__ out,
                           long long n4)
{
    long long i = (long long)blockIdx.x * blockDim.x + threadIdx.x;
    const long long stride = (long long)gridDim.x * blockDim.x;
    for (; i < n4; i += stride) out[i] = in[i];
}

// ---------------------------------------------------------------------------
// Host launcher
// ---------------------------------------------------------------------------
extern "C" void kernel_launch(void* const* d_in, const int* in_sizes, int n_in,
                              void* d_out, int out_size, void* d_ws, size_t ws_size,
                              hipStream_t stream)
{
    (void)in_sizes; (void)n_in; (void)out_size; (void)ws_size;

    const long long Bn = 2, Nq = 4096, Tt = 512, D = 1536, F = 6144;
    const long long MN = Bn * Nq;     // 8192 image tokens
    const long long MT = Bn * Tt;     // 1024 text tokens

    // ---- inputs (setup_inputs order) ----
    const float* x     = (const float*)d_in[0];
    const float* text  = (const float*)d_in[1];
    const float* sq_w  = (const float*)d_in[2];
    const float* sk_w  = (const float*)d_in[3];
    const float* cq_w  = (const float*)d_in[4];
    const float* ck_w  = (const float*)d_in[5];
    const float* ln_w  = (const float*)d_in[6];
    const float* ln_b  = (const float*)d_in[7];
    const float* Wq = (const float*)d_in[8];  const float* bq = (const float*)d_in[9];
    const float* Wk = (const float*)d_in[10]; const float* bk = (const float*)d_in[11];
    const float* Wv = (const float*)d_in[12]; const float* bv = (const float*)d_in[13];
    const float* Wo = (const float*)d_in[14]; const float* bo = (const float*)d_in[15];
    const float* CWq = (const float*)d_in[16]; const float* Cbq = (const float*)d_in[17];
    const float* CWk = (const float*)d_in[18]; const float* Cbk = (const float*)d_in[19];
    const float* CWv = (const float*)d_in[20]; const float* Cbv = (const float*)d_in[21];
    const float* CWo = (const float*)d_in[22]; const float* Cbo = (const float*)d_in[23];
    const float* W1 = (const float*)d_in[24]; const float* b1 = (const float*)d_in[25];
    const float* W2 = (const float*)d_in[26]; const float* b2 = (const float*)d_in[27];

    float* out = (float*)d_out;

    // ---- workspace bump allocator ----
    char*  base = (char*)d_ws;
    size_t off  = 0;
    auto allocB = [&](long long elems) -> bf16* {
        bf16* p = (bf16*)(base + off);
        off += ((size_t)elems * 2 + 255) & ~(size_t)255;
        return p;
    };
    auto allocF = [&](long long elems) -> float* {
        float* p = (float*)(base + off);
        off += ((size_t)elems * 4 + 255) & ~(size_t)255;
        return p;
    };

    // bf16 weight copies, TRANSPOSED to (out_dim x in_dim) row-major
    bf16* wqT  = allocB(D * D); bf16* wkT  = allocB(D * D);
    bf16* wvT  = allocB(D * D); bf16* woT  = allocB(D * D);
    bf16* cwqT = allocB(D * D); bf16* cwkT = allocB(D * D);
    bf16* cwvT = allocB(D * D); bf16* cwoT = allocB(D * D);
    bf16* w1T  = allocB(D * F);   // F x D
    bf16* w2T  = allocB(F * D);   // D x F
    // normalized / cast activations (bf16)
    bf16* a_sq = allocB(MN * D); bf16* a_sk = allocB(MN * D);
    bf16* a_x  = allocB(MN * D); bf16* a_cq = allocB(MN * D);
    bf16* a_ln = allocB(MN * D);
    bf16* t_ck = allocB(MT * D); bf16* t_x  = allocB(MT * D);
    // projections
    bf16* Qb  = allocB(MN * D); bf16* Kb  = allocB(MN * D); bf16* Vb = allocB(MN * D);
    bf16* CQb = allocB(MN * D); bf16* CKb = allocB(MT * D); bf16* CVb = allocB(MT * D);
    // attention scratch
    float* S   = allocF(Bn * Nq * Nq);   bf16* Pb  = allocB(Bn * Nq * Nq);
    float* CS  = allocF(Bn * Nq * Tt);   bf16* CPb = allocB(Bn * Nq * Tt);
    bf16* AOb = allocB(MN * D);
    bf16* CAb = allocB(MN * D);
    bf16* Hb  = allocB(MN * F);

    const float scale = 1.0f / sqrtf((float)D);

    auto cast = [&](const float* src, bf16* dst, long long n) {
        long long g = (n + 255) / 256; if (g > 4096) g = 4096;
        k_cast_bf16<<<(unsigned)g, 256, 0, stream>>>(src, dst, n);
    };
    auto castT = [&](const float* src, bf16* dst, long long K, long long N) {
        long long g = (K * N + 255) / 256; if (g > 4096) g = 4096;
        k_cast_transpose_bf16<<<(unsigned)g, 256, 0, stream>>>(src, dst, (int)K, (int)N);
    };
    auto gemm = [&](const bf16* A, int lda, long long sA,
                    const bf16* Bm, int ldb, long long sB,
                    const float* bias, float* oF, bf16* oB,
                    int ldc, long long sC,
                    long long M, long long Ng, long long K,
                    int batches, int transB, int mode) {
        dim3 g((unsigned)(Ng / 256), (unsigned)(M / 128), (unsigned)batches);
        if (transB)
            k_gemm_bf16_wmma<1><<<g, 256, 0, stream>>>(A, lda, sA, Bm, ldb, sB, bias,
                                                       oF, oB, ldc, sC, (int)K, mode);
        else
            k_gemm_bf16_wmma<0><<<g, 256, 0, stream>>>(A, lda, sA, Bm, ldb, sB, bias,
                                                       oF, oB, ldc, sC, (int)K, mode);
    };

    // ---- stage 0: precision conversion (+weight transpose) + norms ----
    castT(Wq, wqT, D, D);   castT(Wk, wkT, D, D);
    castT(Wv, wvT, D, D);   castT(Wo, woT, D, D);
    castT(CWq, cwqT, D, D); castT(CWk, cwkT, D, D);
    castT(CWv, cwvT, D, D); castT(CWo, cwoT, D, D);
    castT(W1, w1T, D, F);   castT(W2, w2T, F, D);
    cast(x, a_x, MN * D);   cast(text, t_x, MT * D);

    k_rmsnorm_bf16<<<(unsigned)MN, 256, 0, stream>>>(x, sq_w, a_sq, (int)D);
    k_rmsnorm_bf16<<<(unsigned)MN, 256, 0, stream>>>(x, sk_w, a_sk, (int)D);
    k_rmsnorm_bf16<<<(unsigned)MN, 256, 0, stream>>>(x, cq_w, a_cq, (int)D);
    k_rmsnorm_bf16<<<(unsigned)MT, 256, 0, stream>>>(text, ck_w, t_ck, (int)D);
    k_layernorm_bf16<<<(unsigned)MN, 256, 0, stream>>>(x, ln_w, ln_b, a_ln, (int)D);

    // residual init: out = x
    k_copy_f32<<<4096, 256, 0, stream>>>((const float4*)x, (float4*)out, (MN * D) / 4);

    // ---- self attention (weight GEMMs use transB=1 with transposed weights) ----
    gemm(a_sq, (int)D, 0, wqT, (int)D, 0, bq, nullptr, Qb, (int)D, 0, MN, D, D, 1, 1, 1);
    gemm(a_sk, (int)D, 0, wkT, (int)D, 0, bk, nullptr, Kb, (int)D, 0, MN, D, D, 1, 1, 1);
    gemm(a_x,  (int)D, 0, wvT, (int)D, 0, bv, nullptr, Vb, (int)D, 0, MN, D, D, 1, 1, 1);
    // S = Q @ K^T  (per batch)
    gemm(Qb, (int)D, Nq * D, Kb, (int)D, Nq * D, nullptr, S, nullptr,
         (int)Nq, Nq * Nq, Nq, Nq, D, 2, 1, 0);
    k_softmax_bf16<<<dim3((unsigned)Nq, 2), 256, 0, stream>>>(S, Pb, (int)Nq, scale);
    // AO = P @ V (per batch), bf16
    gemm(Pb, (int)Nq, Nq * Nq, Vb, (int)D, Nq * D, nullptr, nullptr, AOb,
         (int)D, Nq * D, Nq, D, Nq, 2, 0, 1);
    // out += AO @ Wo + bo
    gemm(AOb, (int)D, 0, woT, (int)D, 0, bo, out, nullptr, (int)D, 0, MN, D, D, 1, 1, 3);

    // ---- cross attention ----
    gemm(a_cq, (int)D, 0, cwqT, (int)D, 0, Cbq, nullptr, CQb, (int)D, 0, MN, D, D, 1, 1, 1);
    gemm(t_ck, (int)D, 0, cwkT, (int)D, 0, Cbk, nullptr, CKb, (int)D, 0, MT, D, D, 1, 1, 1);
    gemm(t_x,  (int)D, 0, cwvT, (int)D, 0, Cbv, nullptr, CVb, (int)D, 0, MT, D, D, 1, 1, 1);
    // CS = CQ @ CK^T (per batch)
    gemm(CQb, (int)D, Nq * D, CKb, (int)D, Tt * D, nullptr, CS, nullptr,
         (int)Tt, Nq * Tt, Nq, Tt, D, 2, 1, 0);
    k_softmax_bf16<<<dim3((unsigned)Nq, 2), 256, 0, stream>>>(CS, CPb, (int)Tt, scale);
    // CA = CP @ CV (per batch)
    gemm(CPb, (int)Tt, Nq * Tt, CVb, (int)D, Tt * D, nullptr, nullptr, CAb,
         (int)D, Nq * D, Nq, D, Tt, 2, 0, 1);
    // out += CA @ CWo + Cbo
    gemm(CAb, (int)D, 0, cwoT, (int)D, 0, Cbo, out, nullptr, (int)D, 0, MN, D, D, 1, 1, 3);

    // ---- FFN ----
    // H = gelu(ln(x) @ W1 + b1) -> bf16 (fused epilogue)
    gemm(a_ln, (int)D, 0, w1T, (int)D, 0, b1, nullptr, Hb, (int)F, 0, MN, F, D, 1, 1, 2);
    // out += H @ W2 + b2
    gemm(Hb, (int)F, 0, w2T, (int)F, 0, b2, out, nullptr, (int)D, 0, MN, D, F, 1, 1, 3);
}
